// BertSimSelfAttention_34505767256980
// MI455X (gfx1250) — matrix operand order repaired and checked by reading, not compiled, not run on hardware
//
#include <hip/hip_runtime.h>

typedef __bf16 v16bf __attribute__((ext_vector_type(16)));
typedef __bf16 v8bf  __attribute__((ext_vector_type(8)));
typedef float  v8f   __attribute__((ext_vector_type(8)));

#define MM   64
#define SS   256
#define DD   1024
#define HH   16
#define HDD  64
#define XLD  (SS*DD)      // row stride of X within hidden_states
#define WN   (DD*DD)      // elements per weight matrix

static __device__ __forceinline__ v8f zero8() {
    v8f z;
#pragma unroll
    for (int i = 0; i < 8; ++i) z[i] = 0.0f;
    return z;
}

static __device__ __forceinline__ v8f wmma_bf16(v16bf a, v16bf b, v8f c) {
    return __builtin_amdgcn_wmma_f32_16x16x32_bf16(
        false, a, false, b, (short)0, c, false, false);
}

// A fragment (16x32 bf16 from fp32 global), row-major, stride ld.
// Lane l<16: row=l, K halves {0..7,16..23}; lane>=16: row=l-16, {8..15,24..31}.
static __device__ __forceinline__ v16bf load_a_g(const float* base, int ld, int lane) {
    int r  = lane & 15;
    int kb = (lane >> 4) << 3;
    const float* p = base + (size_t)r * ld + kb;
    float4 x0 = *(const float4*)(p + 0);
    float4 x1 = *(const float4*)(p + 4);
    float4 x2 = *(const float4*)(p + 16);
    float4 x3 = *(const float4*)(p + 20);
    v16bf a;
    a[0]  = (__bf16)x0.x; a[1]  = (__bf16)x0.y; a[2]  = (__bf16)x0.z; a[3]  = (__bf16)x0.w;
    a[4]  = (__bf16)x1.x; a[5]  = (__bf16)x1.y; a[6]  = (__bf16)x1.z; a[7]  = (__bf16)x1.w;
    a[8]  = (__bf16)x2.x; a[9]  = (__bf16)x2.y; a[10] = (__bf16)x2.z; a[11] = (__bf16)x2.w;
    a[12] = (__bf16)x3.x; a[13] = (__bf16)x3.y; a[14] = (__bf16)x3.z; a[15] = (__bf16)x3.w;
    return a;
}

// B fragment (32x16) for NT GEMM from pre-converted bf16 weights (row n = B col n).
static __device__ __forceinline__ v16bf load_b_gbf(const __bf16* base, int lane) {
    int n  = lane & 15;
    int kk = (lane >> 4) << 3;
    const __bf16* p = base + (size_t)n * DD + kk;
    v8bf lo = *(const v8bf*)(p);
    v8bf hi = *(const v8bf*)(p + 16);
    v16bf b;
#pragma unroll
    for (int i = 0; i < 8; ++i) { b[i] = lo[i]; b[8 + i] = hi[i]; }
    return b;
}

// A fragment from LDS bf16 row-major [rows][64], K-step ks selects 32-wide block.
static __device__ __forceinline__ v16bf load_a_s(const __bf16* base, int lane, int ks) {
    int r  = lane & 15;
    int kb = ks * 32 + ((lane >> 4) << 3);
    const __bf16* p = base + r * 64 + kb;
    v8bf lo = *(const v8bf*)(p);
    v8bf hi = *(const v8bf*)(p + 16);
    v16bf a;
#pragma unroll
    for (int i = 0; i < 8; ++i) { a[i] = lo[i]; a[8 + i] = hi[i]; }
    return a;
}

// B fragment from LDS bf16 where row n holds B-column n contiguous over K.
static __device__ __forceinline__ v16bf load_b_s(const __bf16* base, int lane, int ks) {
    int n  = lane & 15;
    int kk = ks * 32 + ((lane >> 4) << 3);
    const __bf16* p = base + n * 64 + kk;
    v8bf lo = *(const v8bf*)(p);
    v8bf hi = *(const v8bf*)(p + 16);
    v16bf b;
#pragma unroll
    for (int i = 0; i < 8; ++i) { b[i] = lo[i]; b[8 + i] = hi[i]; }
    return b;
}

// One head of one projection. A-fragments are register-resident (same for all
// heads/projections). W is pre-converted bf16. Writes 64x64 bf16 tile to LDS.
static __device__ __forceinline__ void proj_head(
    const v16bf* a_frags, const __bf16* __restrict__ Wbf,
    const float* __restrict__ bias, int h, __bf16* dst, bool transpose,
    int lane, int wave)
{
    int t0  = wave * 2;
    int tr  = t0 >> 2;
    int tc0 = t0 & 3, tc1 = tc0 + 1;
    (void)tr;

    const __bf16* W0 = Wbf + (size_t)(h * HDD + tc0 * 16) * DD;
    const __bf16* W1 = W0 + (size_t)16 * DD;

    v8f c0 = zero8(), c1 = zero8();
#pragma unroll
    for (int ks = 0; ks < 32; ++ks) {
        v16bf b0 = load_b_gbf(W0 + ks * 32, lane);
        v16bf b1 = load_b_gbf(W1 + ks * 32, lane);
        c0 = wmma_bf16(a_frags[ks], b0, c0);
        c1 = wmma_bf16(a_frags[ks], b1, c1);
    }

    int cn0 = (lane & 15) + tc0 * 16;
    int cn1 = cn0 + 16;
    int r0  = ((lane >> 4) << 3) + ((t0 >> 2) * 16);
    float bb0 = bias[h * HDD + cn0];
    float bb1 = bias[h * HDD + cn1];

    if (!transpose) {
#pragma unroll
        for (int v = 0; v < 8; ++v) {
            dst[(r0 + v) * 64 + cn0] = (__bf16)(c0[v] + bb0);
            dst[(r0 + v) * 64 + cn1] = (__bf16)(c1[v] + bb1);
        }
    } else {
#pragma unroll
        for (int v = 0; v < 8; ++v) {
            dst[cn0 * 64 + r0 + v] = (__bf16)(c0[v] + bb0);
            dst[cn1 * 64 + r0 + v] = (__bf16)(c1[v] + bb1);
        }
    }
}

// ---- prep kernel: fp32 weights -> bf16 workspace (Wq | Wk | Wv) ----
__global__ __launch_bounds__(256)
void convert_weights(const float* __restrict__ Wq, const float* __restrict__ Wk,
                     const float* __restrict__ Wv, __bf16* __restrict__ wbf)
{
    size_t g = ((size_t)blockIdx.x * 256 + threadIdx.x) * 8;
    if (g >= (size_t)3 * WN) return;
    const float* W; size_t off;
    if (g < (size_t)WN)          { W = Wq; off = g; }
    else if (g < (size_t)2 * WN) { W = Wk; off = g - WN; }
    else                         { W = Wv; off = g - (size_t)2 * WN; }
    float4 a = *(const float4*)(W + off);
    float4 b = *(const float4*)(W + off + 4);
    v8bf o;
    o[0] = (__bf16)a.x; o[1] = (__bf16)a.y; o[2] = (__bf16)a.z; o[3] = (__bf16)a.w;
    o[4] = (__bf16)b.x; o[5] = (__bf16)b.y; o[6] = (__bf16)b.z; o[7] = (__bf16)b.w;
    *(v8bf*)(wbf + g) = o;
}

// ---- main fused attention kernel: one block per (b, s) sample ----
__global__ __launch_bounds__(256)
void bert_sim_attn_kernel(const float* __restrict__ hs, const float* __restrict__ am,
                          const float* __restrict__ sim,
                          const __bf16* __restrict__ wbf,
                          const float* __restrict__ bq, const float* __restrict__ bk,
                          const float* __restrict__ bv,
                          float* __restrict__ out)
{
    __shared__ __align__(16) __bf16 q_s[64 * 64];
    __shared__ __align__(16) __bf16 k_s[64 * 64];
    __shared__ __align__(16) __bf16 vT_s[64 * 64];   // V^T: [hd][m]
    __shared__ __align__(16) __bf16 p_s[64 * 64];    // probs bf16
    __shared__ float s_s[64 * 65];                   // scores, row-padded
    __shared__ float mbias[64];
    __shared__ float red[64 * 4];
    __shared__ float rsum[64 * 4];

    int bs  = blockIdx.x;
    int b   = bs >> 8;
    int sq  = bs & 255;
    int tid = threadIdx.x, lane = tid & 31, wave = tid >> 5;

    const float* X    = hs  + (size_t)b * MM * SS * DD + (size_t)sq * DD;
    const float* simb = sim + (size_t)bs * MM * MM;
    const __bf16* Wq_bf = wbf;
    const __bf16* Wk_bf = wbf + (size_t)WN;
    const __bf16* Wv_bf = wbf + (size_t)2 * WN;

    int t0  = wave * 2;
    int tr  = t0 >> 2;
    int tc0 = t0 & 3, tc1 = tc0 + 1;
    int cn0 = (lane & 15) + tc0 * 16;
    int cn1 = cn0 + 16;
    int r0  = ((lane >> 4) << 3) + tr * 16;

    // Register-resident A operand: this wave's 16-row X band, all K, bf16.
    // Converted exactly once; reused for all heads and all three projections.
    v16bf a_frags[32];
    {
        const float* Xr = X + (size_t)tr * 16 * XLD;
#pragma unroll
        for (int ks = 0; ks < 32; ++ks)
            a_frags[ks] = load_a_g(Xr + ks * 32, XLD, lane);
    }

    if (tid < 64) {
        float amv = am[((size_t)b * MM + tid) * SS + sq];
        mbias[tid] = (1.0f - amv) * -10000.0f;
    }

    int srow = tid >> 2, spart = tid & 3, sc0 = spart * 16;

    for (int h = 0; h < HH; ++h) {
        __syncthreads();   // LDS reuse fence (also covers mbias at h=0)
        proj_head(a_frags, Wq_bf, bq, h, q_s,  false, lane, wave);
        proj_head(a_frags, Wk_bf, bk, h, k_s,  false, lane, wave);
        proj_head(a_frags, Wv_bf, bv, h, vT_s, true,  lane, wave);
        __syncthreads();

        // S = (q * k^T)/8 * sim + key mask
        {
            v8f c0 = zero8(), c1 = zero8();
#pragma unroll
            for (int ks = 0; ks < 2; ++ks) {
                v16bf a  = load_a_s(q_s + tr * 16 * 64, lane, ks);
                v16bf b0 = load_b_s(k_s + tc0 * 16 * 64, lane, ks);
                v16bf b1 = load_b_s(k_s + tc1 * 16 * 64, lane, ks);
                c0 = wmma_bf16(a, b0, c0);
                c1 = wmma_bf16(a, b1, c1);
            }
#pragma unroll
            for (int v = 0; v < 8; ++v) {
                int r = r0 + v;
                s_s[r * 65 + cn0] = c0[v] * 0.125f * simb[r * MM + cn0] + mbias[cn0];
                s_s[r * 65 + cn1] = c1[v] * 0.125f * simb[r * MM + cn1] + mbias[cn1];
            }
        }
        __syncthreads();

        // softmax over key axis: 4 threads per row, 16 cols each
        {
            float mx = -3.4e38f;
#pragma unroll
            for (int c = 0; c < 16; ++c)
                mx = fmaxf(mx, s_s[srow * 65 + sc0 + c]);
            red[tid] = mx;
        }
        __syncthreads();
        {
            float gmx = fmaxf(fmaxf(red[srow * 4 + 0], red[srow * 4 + 1]),
                              fmaxf(red[srow * 4 + 2], red[srow * 4 + 3]));
            float sum = 0.0f;
#pragma unroll
            for (int c = 0; c < 16; ++c) {
                float e = __expf(s_s[srow * 65 + sc0 + c] - gmx);
                s_s[srow * 65 + sc0 + c] = e;
                sum += e;
            }
            rsum[tid] = sum;
        }
        __syncthreads();
        {
            float gs = (rsum[srow * 4 + 0] + rsum[srow * 4 + 1]) +
                       (rsum[srow * 4 + 2] + rsum[srow * 4 + 3]);
            float inv = 1.0f / gs;
#pragma unroll
            for (int c = 0; c < 16; ++c)
                p_s[srow * 64 + sc0 + c] = (__bf16)(s_s[srow * 65 + sc0 + c] * inv);
        }
        __syncthreads();

        // ctx = P * V  (vT_s row n = V column n -> contiguous K)
        {
            v8f d0 = zero8(), d1 = zero8();
#pragma unroll
            for (int ks = 0; ks < 2; ++ks) {
                v16bf a  = load_a_s(p_s + tr * 16 * 64, lane, ks);
                v16bf b0 = load_b_s(vT_s + tc0 * 16 * 64, lane, ks);
                v16bf b1 = load_b_s(vT_s + tc1 * 16 * 64, lane, ks);
                d0 = wmma_bf16(a, b0, d0);
                d1 = wmma_bf16(a, b1, d1);
            }
            float* ob = out + (size_t)bs * MM * DD + (size_t)h * HDD;
#pragma unroll
            for (int v = 0; v < 8; ++v) {
                ob[(size_t)(r0 + v) * DD + cn0] = d0[v];
                ob[(size_t)(r0 + v) * DD + cn1] = d1[v];
            }
        }
    }
}

extern "C" void kernel_launch(void* const* d_in, const int* in_sizes, int n_in,
                              void* d_out, int out_size, void* d_ws, size_t ws_size,
                              hipStream_t stream) {
    const float* hs  = (const float*)d_in[0];
    const float* am  = (const float*)d_in[1];
    const float* sim = (const float*)d_in[2];
    const float* Wq  = (const float*)d_in[3];
    const float* bq  = (const float*)d_in[4];
    const float* Wk  = (const float*)d_in[5];
    const float* bk  = (const float*)d_in[6];
    const float* Wv  = (const float*)d_in[7];
    const float* bv  = (const float*)d_in[8];
    (void)in_sizes; (void)n_in; (void)out_size; (void)ws_size;

    __bf16* wbf = (__bf16*)d_ws;   // 3 * 1024 * 1024 bf16 = 6 MB scratch

    int convBlocks = (3 * WN) / (256 * 8);   // 1536
    convert_weights<<<dim3(convBlocks), dim3(256), 0, stream>>>(Wq, Wk, Wv, wbf);

    bert_sim_attn_kernel<<<dim3(4 * SS), dim3(256), 0, stream>>>(
        hs, am, sim, wbf, bq, bk, bv, (float*)d_out);
}
